// CTO_stitchvit_76003741270284
// MI455X (gfx1250) — matrix operand order repaired and checked by reading, not compiled, not run on hardware
//
#include <hip/hip_runtime.h>
#include <cmath>

// ---------- CDNA5 WMMA types ----------
typedef __attribute__((ext_vector_type(16))) __bf16          v16bf;
typedef __attribute__((ext_vector_type(16))) unsigned short  v16us;
typedef __attribute__((ext_vector_type(8)))  unsigned short  v8us;
typedef __attribute__((ext_vector_type(8)))  unsigned int    v8u;
typedef __attribute__((ext_vector_type(4)))  unsigned int    v4u;
typedef __attribute__((ext_vector_type(8)))  float           v8f;
typedef __attribute__((ext_vector_type(4)))  float           v4f;

#define DEVINL static __device__ __forceinline__

constexpr int kB  = 8;
constexpr int kC  = 256;
constexpr int kH  = 128;
constexpr int kW  = 128;
constexpr int kHW = kH * kW;            // 16384
constexpr int kDK = 64;                 // channels per attention group
// token-major intermediate layout: [g][b][l][d]; one (g,b) block = 2^20 elems

// f32 -> bf16 bits (round to nearest even)
DEVINL unsigned short f2bf(float f) {
  unsigned int u = __builtin_bit_cast(unsigned int, f);
  unsigned int r = u + 0x7fffu + ((u >> 16) & 1u);
  return (unsigned short)(r >> 16);
}

DEVINL unsigned int pack2(float lo, float hi) {
#if __has_builtin(__builtin_amdgcn_cvt_pk_bf16_f32)
  auto p = __builtin_amdgcn_cvt_pk_bf16_f32(lo, hi);
  return __builtin_bit_cast(unsigned int, p);
#else
  return (unsigned int)f2bf(lo) | ((unsigned int)f2bf(hi) << 16);
#endif
}

DEVINL v16us cat8(v8us a, v8us b) {
  return __builtin_shufflevector(a, b, 0, 1, 2, 3, 4, 5, 6, 7,
                                 8, 9, 10, 11, 12, 13, 14, 15);
}

DEVINL v8f wmma_bf16(v16us a, v16us b, v8f c) {
  return __builtin_amdgcn_wmma_f32_16x16x32_bf16(
      false, __builtin_bit_cast(v16bf, a),
      false, __builtin_bit_cast(v16bf, b),
      (short)0, c, false, false);
}

// 16x32 bf16 A fragment from pre-converted bf16 row: K runs [16h..16h+15];
// caller passes p = rowbase + 16h. Pure b128 loads, no VALU.
DEVINL v16us afrag_bf16(const unsigned short* __restrict__ p) {
  return cat8(*(const v8us*)p, *(const v8us*)(p + 8));
}

// =====================================================================
// Kernel 0: one-shot f32 -> bf16 conversion (x and all weights), so no
// conversion VALU remains inside GEMM inner loops. n multiple of 8.
// =====================================================================
__global__ __launch_bounds__(256) void cvt_f32_bf16(
    const float* __restrict__ src, unsigned short* __restrict__ dst, int n) {
  const int i8 = (blockIdx.x * 256 + threadIdx.x) * 8;
  if (i8 >= n) return;
  v4f a = *(const v4f*)(src + i8);
  v4f b = *(const v4f*)(src + i8 + 4);
  v4u w;
  w[0] = pack2(a[0], a[1]); w[1] = pack2(a[2], a[3]);
  w[2] = pack2(b[0], b[1]); w[3] = pack2(b[2], b[3]);
  *(v8us*)(dst + i8) = __builtin_bit_cast(v8us, w);
}

// =====================================================================
// Kernel 1: fused Q/K/V 1x1 conv GEMM (all-bf16 operands); writes
// token-major bf16. grid (HW/64, B, 3), block 256 (8 waves).
// =====================================================================
__global__ __launch_bounds__(256) void qkv_gemm(
    const unsigned short* __restrict__ xbf,
    const unsigned short* __restrict__ Wqb, const float* __restrict__ bq,
    const unsigned short* __restrict__ Wkb, const float* __restrict__ bk,
    const unsigned short* __restrict__ Wvb, const float* __restrict__ bv,
    unsigned short* __restrict__ qt, unsigned short* __restrict__ kt,
    unsigned short* __restrict__ vt) {
  const int n0 = blockIdx.x * 64;
  const int bb = blockIdx.y;
  const int which = blockIdx.z;
  const unsigned short* Wg = (which == 0) ? Wqb : (which == 1) ? Wkb : Wvb;
  const float* bg = (which == 0) ? bq : (which == 1) ? bk : bv;
  unsigned short* og = (which == 0) ? qt : (which == 1) ? kt : vt;

  __shared__ unsigned short Bs[64][40];   // [n][k] bf16, 20-bank row stride

  const int t = threadIdx.x;
  const int lane = t & 31, wv = t >> 5, h = lane >> 4, ln = lane & 15;

  v8f acc[2][4] = {};
  const unsigned short* xb = xbf + (size_t)bb * kC * kHW;

  for (int k0 = 0; k0 < kC; k0 += 32) {
    {  // stage 32K x 64N: b128 bf16 loads, b16 transpose stores (no packing)
      const int k = t >> 3;               // 0..31
      const int n8 = (t & 7) * 8;
      const v8us r = *(const v8us*)(xb + (size_t)(k0 + k) * kHW + n0 + n8);
#pragma unroll
      for (int u = 0; u < 8; ++u) Bs[n8 + u][k] = r[u];
    }
    __syncthreads();

    v16us bfrag[4];
#pragma unroll
    for (int nt = 0; nt < 4; ++nt)
      bfrag[nt] = cat8(*(const v8us*)&Bs[nt * 16 + ln][h * 16],
                       *(const v8us*)&Bs[nt * 16 + ln][h * 16 + 8]);

#pragma unroll
    for (int mt = 0; mt < 2; ++mt) {
      const int m = wv * 32 + mt * 16 + ln;
      const v16us au = afrag_bf16(Wg + (size_t)m * kC + k0 + 16 * h);
#pragma unroll
      for (int nt = 0; nt < 4; ++nt)
        acc[mt][nt] = wmma_bf16(au, bfrag[nt], acc[mt][nt]);
    }
    __syncthreads();
  }

  // epilogue: + bias, scatter to token-major [g][b][l][d] (pow2 shifts only)
  const int gI = wv >> 1;                  // wave's 32 rows sit in one group
  const int ss = 4 - gI;                   // log2(hs)
  const int hsm1 = (1 << ss) - 1;
  const int log2ow = 7 - ss, log2ohow = 14 - 2 * ss, log2D = 20 - 2 * ss;
  unsigned short* od = og + (((size_t)gI * kB + bb) << 20);
  const int y = n0 >> 7;                   // constant per block
  const int xb0 = n0 & 127;
  const int ii = y & hsm1, oi = y >> ss;
  size_t pre[4];                           // hoisted (ltok<<log2D)+oj per nt
#pragma unroll
  for (int nt = 0; nt < 4; ++nt) {
    const int xx = xb0 + nt * 16 + ln;
    const int jj = xx & hsm1, oj = xx >> ss;
    pre[nt] = ((size_t)((ii << ss) | jj) << log2D) + (size_t)oj;
  }
#pragma unroll
  for (int mt = 0; mt < 2; ++mt) {
#pragma unroll
    for (int j = 0; j < 8; ++j) {
      const int m = wv * 32 + mt * 16 + 8 * h + j;
      const float bias = bg[m];
      const size_t rowb = ((size_t)(m & 63) << log2ohow) + ((size_t)oi << log2ow);
#pragma unroll
      for (int nt = 0; nt < 4; ++nt)
        od[rowb + pre[nt]] = f2bf(acc[mt][nt][j] + bias);
    }
  }
}

// =====================================================================
// Kernel 2a: split-K scores S[l,m] += sum_d qt[l,d]*kt[m,d].
// All fragments are contiguous b128 loads; f32 atomics accumulate.
// grid (D/2048, LT*LT, B).
// =====================================================================
__global__ __launch_bounds__(256) void attn_scores(
    const unsigned short* __restrict__ qt, const unsigned short* __restrict__ kt,
    float* __restrict__ S, int g, int L, int LT, int Dlen) {
  const int bb = blockIdx.z;
  const int lt = blockIdx.y % LT, mt = blockIdx.y / LT;
  const int lane = threadIdx.x & 31, wv = threadIdx.x >> 5;
  const int h = lane >> 4, ln = lane & 15;

  const size_t gb = (((size_t)g * kB + bb) << 20);
  const int l = lt * 16 + ln;
  const int m = mt * 16 + ln;
  const unsigned short* qrow = qt + gb + (size_t)l * Dlen;
  const unsigned short* krow = kt + gb + (size_t)m * Dlen;
  const bool lv = (l < L), mv = (m < L);
  const v16us zero16 = {};

  v8f acc = {};
  const int kBase0 = blockIdx.x * 2048 + wv * 256;
#pragma unroll
  for (int step = 0; step < 8; ++step) {
    const int kb0 = kBase0 + step * 32;
    const v16us au = lv ? afrag_bf16(qrow + kb0 + 16 * h) : zero16;
    const v16us bu = mv ? cat8(*(const v8us*)(krow + kb0 + 16 * h),
                               *(const v8us*)(krow + kb0 + 16 * h + 8))
                        : zero16;
    acc = wmma_bf16(au, bu, acc);
  }

#pragma unroll
  for (int j = 0; j < 8; ++j) {
    const int row = lt * 16 + 8 * h + j;
    const int col = mt * 16 + ln;
    if (row < L && col < L)
      atomicAdd(&S[((size_t)bb * L + row) * L + col], acc[j]);
  }
}

// =====================================================================
// Kernel 2b: row softmax -> zero-padded bf16 P (Lpad x Lpad per batch).
// Padding rows/cols written as zero so attn_pv loads are unguarded.
// grid (Lpad, B), block 256.
// =====================================================================
__global__ __launch_bounds__(256) void softmax_bf16(
    const float* __restrict__ S, unsigned short* __restrict__ Pb,
    int L, int Lpad, float scale) {
  const int row = blockIdx.x, bb = blockIdx.y;
  const int t = threadIdx.x;
  unsigned short* prow = Pb + ((size_t)bb * Lpad + row) * Lpad;
  if (row >= L) {                          // block-uniform branch
    if (t < Lpad) prow[t] = 0;
    return;
  }
  const float* p = S + ((size_t)bb * L + row) * L;
  __shared__ float red[256];

  const float v = (t < L) ? p[t] * scale : -3.4e38f;
  red[t] = v;
  __syncthreads();
  for (int off = 128; off > 0; off >>= 1) {
    if (t < off) red[t] = fmaxf(red[t], red[t + off]);
    __syncthreads();
  }
  const float mx = red[0];
  __syncthreads();

  const float e = (t < L) ? __expf(v - mx) : 0.0f;
  red[t] = e;
  __syncthreads();
  for (int off = 128; off > 0; off >>= 1) {
    if (t < off) red[t] += red[t + off];
    __syncthreads();
  }
  const float inv = 1.0f / red[0];
  if (t < Lpad) prow[t] = (t < L) ? f2bf(e * inv) : (unsigned short)0;
}

// =====================================================================
// Kernel 2c: Y[l,d] = sum_tok P[l,tok]*vt[tok,d]; token-major in/out.
// Block stages 32(tok) x 128(d) V tile transposed in LDS; 8 waves own
// 16-wide d slices. A fragments: unguarded b128 loads from padded bf16 P.
// grid (LT * D/128, B), block 256.
// =====================================================================
__global__ __launch_bounds__(256) void attn_pv(
    const unsigned short* __restrict__ Pb, const unsigned short* __restrict__ vt,
    unsigned short* __restrict__ yt, int g, int L, int LT, int Lpad, int Dlen) {
  const int bb = blockIdx.y;
  const int lt = blockIdx.x % LT, dblk = blockIdx.x / LT;
  __shared__ unsigned short Vs[128][40];

  const int t = threadIdx.x;
  const int lane = t & 31, wv = t >> 5, h = lane >> 4, ln = lane & 15;

  const size_t gb = (((size_t)g * kB + bb) << 20);
  const unsigned short* vg = vt + gb;
  unsigned short* yg = yt + gb;
  const unsigned short* prow =
      Pb + ((size_t)bb * Lpad + lt * 16 + ln) * Lpad;   // rows < Lpad always
  const int d0 = dblk * 128;

  v8f acc = {};
  const int nk = (L + 31) / 32;
  for (int ks = 0; ks < nk; ++ks) {
    const int kb0 = ks * 32;
    {  // stage V[tok][d] -> Vs[d][tok] (b128 loads, b16 transpose stores)
      const int tt = t >> 3;               // 0..31
      const int dd = (t & 7) * 16;
      v8us r0 = {}, r1 = {};
      const int tok = kb0 + tt;
      if (tok < L) {
        const unsigned short* vp = vg + (size_t)tok * Dlen + d0 + dd;
        r0 = *(const v8us*)vp;
        r1 = *(const v8us*)(vp + 8);
      }
#pragma unroll
      for (int u = 0; u < 8; ++u) {
        Vs[dd + u][tt] = r0[u];
        Vs[dd + 8 + u][tt] = r1[u];
      }
    }
    __syncthreads();

    const v16us au = afrag_bf16(prow + kb0 + 16 * h);
    const v16us bu = cat8(*(const v8us*)&Vs[wv * 16 + ln][h * 16],
                          *(const v8us*)&Vs[wv * 16 + ln][h * 16 + 8]);
    acc = wmma_bf16(au, bu, acc);
    __syncthreads();
  }

#pragma unroll
  for (int j = 0; j < 8; ++j) {
    const int lr = lt * 16 + 8 * h + j;
    if (lr < L)
      yg[(size_t)lr * Dlen + d0 + wv * 16 + ln] = f2bf(acc[j]);
  }
}

// =====================================================================
// Kernel 3: 3x3 SAME conv as implicit GEMM (K = 2304) over token-major
// attention output + bias + BatchNorm + LeakyReLU(0.2). f32 NCHW out.
// A operand from pre-converted bf16 Wo (contiguous: m*2304 + kk).
// grid (HW/64, B), block 256.
// =====================================================================
__global__ __launch_bounds__(256) void conv3_bn_lrelu(
    const unsigned short* __restrict__ yt, const unsigned short* __restrict__ Wob,
    const float* __restrict__ bo, const float* __restrict__ gamma,
    const float* __restrict__ beta, const float* __restrict__ mean,
    const float* __restrict__ var, float* __restrict__ out) {
  const int n0 = blockIdx.x * 64;
  const int bb = blockIdx.y;
  __shared__ unsigned short Bs[64][40];

  const int t = threadIdx.x;
  const int lane = t & 31, wv = t >> 5, h = lane >> 4, ln = lane & 15;

  v8f acc[2][4] = {};
  const int y = n0 >> 7;                 // constant per block
  const int xb0 = n0 & 127;

  for (int k0 = 0; k0 < kC * 9; k0 += 32) {
    {  // stage im2col tile: token-major gather (shift math), pack k pairs
      const int n = t & 63;
      const int kp = (t >> 6) * 2;       // 0,2,4,6
      const int xx0 = xb0 + n;
#pragma unroll
      for (int i = 0; i < 4; ++i) {
        const int kbase = kp + i * 8;
        unsigned int packed = 0;
#pragma unroll
        for (int half = 0; half < 2; ++half) {
          const int kk = k0 + kbase + half;
          const int c = kk / 9, rs = kk - c * 9;    // const divisors
          const int r = rs / 3, s2 = rs - r * 3;
          const int yy = y + r - 1, xx = xx0 + s2 - 1;
          unsigned int val = 0;
          if ((unsigned)yy < (unsigned)kH && (unsigned)xx < (unsigned)kW) {
            const int gg = c >> 6, cg = c & 63;
            const int ss = 4 - gg;
            const int ii = yy & ((1 << ss) - 1), jj = xx & ((1 << ss) - 1);
            const int oi = yy >> ss, oj = xx >> ss;
            const int ltok = (ii << ss) | jj;
            const size_t off = (((size_t)gg * kB + bb) << 20) +
                               ((size_t)ltok << (20 - 2 * ss)) +
                               ((size_t)cg << (14 - 2 * ss)) +
                               ((size_t)oi << (7 - ss)) + (size_t)oj;
            val = yt[off];
          }
          packed |= val << (16 * half);
        }
        *(unsigned int*)&Bs[n][kbase] = packed;
      }
    }
    __syncthreads();

    v16us bfrag[4];
#pragma unroll
    for (int nt = 0; nt < 4; ++nt)
      bfrag[nt] = cat8(*(const v8us*)&Bs[nt * 16 + ln][h * 16],
                       *(const v8us*)&Bs[nt * 16 + ln][h * 16 + 8]);

#pragma unroll
    for (int mt = 0; mt < 2; ++mt) {
      const int m = wv * 32 + mt * 16 + ln;
      const v16us au = afrag_bf16(Wob + (size_t)m * 2304 + k0 + 16 * h);
#pragma unroll
      for (int nt = 0; nt < 4; ++nt)
        acc[mt][nt] = wmma_bf16(au, bfrag[nt], acc[mt][nt]);
    }
    __syncthreads();
  }

  float* ob = out + (size_t)bb * kC * kHW;
  int nOff[4];
#pragma unroll
  for (int nt = 0; nt < 4; ++nt) nOff[nt] = n0 + nt * 16 + ln;
#pragma unroll
  for (int mt = 0; mt < 2; ++mt) {
#pragma unroll
    for (int j = 0; j < 8; ++j) {
      const int m = wv * 32 + mt * 16 + 8 * h + j;
      const float inv = gamma[m] * rsqrtf(var[m] + 1e-5f);
      const float shift = beta[m] - mean[m] * inv;
      const float bias = bo[m];
#pragma unroll
      for (int nt = 0; nt < 4; ++nt) {
        float v = acc[mt][nt][j] + bias;
        v = v * inv + shift;
        v = (v >= 0.0f) ? v : 0.2f * v;
        ob[(size_t)m * kHW + nOff[nt]] = v;
      }
    }
  }
}

// =====================================================================
// Host launcher
// =====================================================================
extern "C" void kernel_launch(void* const* d_in, const int* in_sizes, int n_in,
                              void* d_out, int out_size, void* d_ws, size_t ws_size,
                              hipStream_t stream) {
  (void)in_sizes; (void)n_in; (void)out_size; (void)ws_size;

  const float* x     = (const float*)d_in[0];
  const float* Wq    = (const float*)d_in[1];
  const float* bq    = (const float*)d_in[2];
  const float* Wk    = (const float*)d_in[3];
  const float* bk    = (const float*)d_in[4];
  const float* Wv    = (const float*)d_in[5];
  const float* bv    = (const float*)d_in[6];
  const float* Wo    = (const float*)d_in[7];
  const float* bo    = (const float*)d_in[8];
  const float* gamma = (const float*)d_in[9];
  const float* beta  = (const float*)d_in[10];
  const float* mean  = (const float*)d_in[11];
  const float* var   = (const float*)d_in[12];
  float* out = (float*)d_out;

  // Workspace: 5 x 64MB bf16 tensors + bf16 weights + padded P + f32 scores
  const size_t T = (size_t)kB * kC * kHW;            // 33,554,432 elems
  unsigned short* ws16 = (unsigned short*)d_ws;
  unsigned short* qbuf = ws16;
  unsigned short* kbuf = qbuf + T;
  unsigned short* vbuf = kbuf + T;
  unsigned short* ybuf = vbuf + T;
  unsigned short* xbf  = ybuf + T;
  unsigned short* Wqb  = xbf + T;
  unsigned short* Wkb  = Wqb + 65536;
  unsigned short* Wvb  = Wkb + 65536;
  unsigned short* Wob  = Wvb + 65536;                // 589,824 elems
  unsigned short* Pb   = Wob + 589824;               // 8*256*256 = 524,288
  float*          S    = (float*)(Pb + 524288);      // 2 MB

  // 0) one-shot bf16 conversions (weights + input)
  cvt_f32_bf16<<<32, 256, 0, stream>>>(Wq, Wqb, 65536);
  cvt_f32_bf16<<<32, 256, 0, stream>>>(Wk, Wkb, 65536);
  cvt_f32_bf16<<<32, 256, 0, stream>>>(Wv, Wvb, 65536);
  cvt_f32_bf16<<<288, 256, 0, stream>>>(Wo, Wob, 589824);
  cvt_f32_bf16<<<16384, 256, 0, stream>>>(x, xbf, (int)T);

  // 1) fused QKV GEMM
  qkv_gemm<<<dim3(kHW / 64, kB, 3), 256, 0, stream>>>(
      xbf, Wqb, bq, Wkb, bk, Wvb, bv, qbuf, kbuf, vbuf);

  // 2) strided attention per group (sequential on stream, shared S/Pb)
  const int strides[4] = {16, 8, 4, 2};
  for (int g = 0; g < 4; ++g) {
    const int hs = strides[g];
    const int L = hs * hs;
    const int LT = (L + 15) / 16;
    const int Lpad = ((L + 31) / 32) * 32;
    const int Dlen = (kDK * kHW) / L;      // 2^20 / L, multiple of 2048

    hipMemsetAsync(S, 0, (size_t)kB * L * L * sizeof(float), stream);

    attn_scores<<<dim3(Dlen / 2048, LT * LT, kB), 256, 0, stream>>>(
        qbuf, kbuf, S, g, L, LT, Dlen);

    const float scale = 1.0f / sqrtf((float)Dlen);
    softmax_bf16<<<dim3(Lpad, kB), 256, 0, stream>>>(S, Pb, L, Lpad, scale);

    attn_pv<<<dim3(LT * (Dlen / 128), kB), 256, 0, stream>>>(
        Pb, vbuf, ybuf, g, L, LT, Lpad, Dlen);
  }

  // 3) 3x3 conv + BN + LeakyReLU
  conv3_bn_lrelu<<<dim3(kHW / 64, kB), 256, 0, stream>>>(
      ybuf, Wob, bo, gamma, beta, mean, var, out);
}
